// WindowAttention_67637144977907
// MI455X (gfx1250) — matrix operand order repaired and checked by reading, not compiled, not run on hardware
//
#include <hip/hip_runtime.h>
#include <math.h>
#include <stdint.h>

// ---------------------------------------------------------------------------
// Swin WindowAttention for MI455X (gfx1250), bf16 WMMA path.
// BS=4096, N=49 (pad 64), C=384, heads=12, hd=32, NW=64, window=7.
// ---------------------------------------------------------------------------

typedef __attribute__((ext_vector_type(16))) __bf16 v16bf_t;
typedef __attribute__((ext_vector_type(8)))  float  v8f_t;

#define BATCH  4096
#define SEQ    49
#define CDIM   384
#define NHEAD  12
#define HD     32
#define NWIN   64
#define WIN    7
#define NP     64          // padded sequence length
#define MROWS  (BATCH*SEQ) // 200704, divisible by 128
#define KD     384         // K depth of both GEMMs

// ---- gfx1250 async global->LDS (ASYNCcnt) path, feature-guarded ------------
#if defined(__AMDGCN__) && defined(__has_builtin)
#  if __has_builtin(__builtin_amdgcn_global_load_async_to_lds_b128)
#    define HAVE_ASYNC_LDS 1
#  endif
#endif
#ifndef HAVE_ASYNC_LDS
#  define HAVE_ASYNC_LDS 0
#endif

#if HAVE_ASYNC_LDS
// Builtin prototype (from hipcc diagnostic): param 1 is int4 in AS1 (__device__).
typedef int v4i_vs_t __attribute__((vector_size(16)));
typedef __attribute__((address_space(1))) v4i_vs_t* g4p_t;   // global int4*
typedef __attribute__((address_space(3))) v4i_vs_t* l4p_t;   // LDS int4*
static __device__ __forceinline__ void async_copy16(const void* g, void* l) {
    // generic->AS1: identical bits; generic LDS ptr low 32 bits = LDS offset
    __builtin_amdgcn_global_load_async_to_lds_b128(
        (g4p_t)(uintptr_t)g, (l4p_t)(unsigned int)(uintptr_t)l, 0, 0);
}
#endif

static __device__ __forceinline__ void async_wait0() {
#if defined(__AMDGCN__)
#  if defined(__has_builtin) && __has_builtin(__builtin_amdgcn_s_wait_asynccnt)
    __builtin_amdgcn_s_wait_asynccnt(0);
#  else
    asm volatile("s_wait_asynccnt 0x0" ::: "memory");
#  endif
#endif
}

static __device__ __forceinline__ unsigned short f2bf(float f) {
    unsigned int u = __builtin_bit_cast(unsigned int, f);
    unsigned int r = (u + 0x7FFFu + ((u >> 16) & 1u)) >> 16;   // RNE
    return (unsigned short)r;
}
static __device__ __forceinline__ __bf16 bfbits(unsigned short s) {
    return __builtin_bit_cast(__bf16, s);
}

// ---------------------------------------------------------------------------
// GEMM + bias: Out[M][NT*128] = A[M][384] * B[384][NT*128] + bias.
// Block owns a 128-row A strip for the FULL K (staged once in 96KB LDS), then
// loops over all N-tiles -> A is read from HBM exactly once. B tile (32x128,
// staged transposed) is double-buffered so fetch+cvt overlaps WMMA.
// Per wave per K-step: 2 A-frags + 4 B-frags shared across 8 WMMAs (2x4 tiles).
// Dynamic LDS: 128*384*2 (A) + 2*128*32*2 (B) = 114688 B (<320KB/WGP).
// ---------------------------------------------------------------------------
#define GEMM_LDS_BYTES ((size_t)(128 * KD + 2 * 128 * 32) * sizeof(unsigned short))

template<bool A_IS_BF16, bool OUT_BF16, int NT>
__global__ __launch_bounds__(256)
void gemm_bias_kernel(const void* __restrict__ Aptr,
                      const float* __restrict__ B,
                      const float* __restrict__ bias,
                      void* __restrict__ Out)
{
    constexpr int NOUT = NT * 128;
    extern __shared__ unsigned short smem[];
    unsigned short* lds_a  = smem;                // [128][384] bf16
    unsigned short* lds_bt = smem + 128 * KD;     // 2 x [128 n][32 k] bf16

    const int tid  = threadIdx.x;
    const int lane = tid & 31;
    const int wave = tid >> 5;
    const int m0   = blockIdx.x * 128;

    // ---- stage A strip 128x384 (once per block) ----
    {
        const int row = tid >> 1;                  // 0..127
        const int cb  = (tid & 1) * (KD / 2);      // 0 or 192
        const size_t base = (size_t)(m0 + row) * KD + (size_t)cb;
        if constexpr (A_IS_BF16) {
            const unsigned short* Ab = (const unsigned short*)Aptr;
#if HAVE_ASYNC_LDS
            #pragma unroll
            for (int c = 0; c < KD / 2; c += 8)    // 16B raw async copies
                async_copy16(Ab + base + c, &lds_a[row * KD + cb + c]);
#else
            for (int c = 0; c < KD / 2; ++c)
                lds_a[row * KD + cb + c] = Ab[base + c];
#endif
        } else {
            const float* Af = (const float*)Aptr;
            for (int c = 0; c < KD / 2; ++c)
                lds_a[row * KD + cb + c] = f2bf(Af[base + c]);
        }
    }
#if HAVE_ASYNC_LDS
    if constexpr (A_IS_BF16) async_wait0();
#endif

    const int brow = tid >> 3;          // 0..31 (k within tile)
    const int bc0  = (tid & 7) * 16;    // 0..112 (n within tile)
    const int rt0  = (wave >> 1) * 2;   // row tiles {rt0, rt0+1}
    const int ct0  = (wave & 1) * 4;    // col tiles ct0..ct0+3
    const int g    = lane >> 4;
    const int hl   = lane & 15;

    for (int nt = 0; nt < NT; ++nt) {
        const int n0 = nt * 128;
        __syncthreads();   // A-strip ready (1st) / retire prev nt's B reads

        // prologue: stage k-step 0 into buffer 0
        {
            const size_t base = (size_t)brow * NOUT + (size_t)(n0 + bc0);
            #pragma unroll
            for (int u = 0; u < 16; ++u)
                lds_bt[(bc0 + u) * 32 + brow] = f2bf(B[base + u]);
        }

        v8f_t acc[2][4] = {{{}, {}, {}, {}}, {{}, {}, {}, {}}};
        int p = 0;
        for (int ks = 0; ks < KD / 32; ++ks) {
            __syncthreads();                       // bt[p] staged by everyone
            if (ks + 1 < KD / 32) {                // stage next into bt[p^1]
                const size_t base =
                    (size_t)(32 * (ks + 1) + brow) * NOUT + (size_t)(n0 + bc0);
                unsigned short* dst = lds_bt + (p ^ 1) * (128 * 32);
                #pragma unroll
                for (int u = 0; u < 16; ++u)
                    dst[(bc0 + u) * 32 + brow] = f2bf(B[base + u]);
            }
            const unsigned short* bt = lds_bt + p * (128 * 32);

            v16bf_t af[2];
            #pragma unroll
            for (int rr = 0; rr < 2; ++rr) {
                const int r = (rt0 + rr) * 16 + hl;
                #pragma unroll
                for (int e = 0; e < 16; ++e) {
                    const int kk = ks * 32 + 16 * g + e;
                    af[rr][e] = bfbits(lds_a[r * KD + kk]);
                }
            }
            #pragma unroll
            for (int t = 0; t < 4; ++t) {
                const int bn = (ct0 + t) * 16 + hl;
                v16bf_t bfrag;
                #pragma unroll
                for (int e = 0; e < 16; ++e)       // contiguous 32B
                    bfrag[e] = bfbits(bt[bn * 32 + 16 * g + e]);
                acc[0][t] = __builtin_amdgcn_wmma_f32_16x16x32_bf16(
                    false, af[0], false, bfrag, (short)0, acc[0][t], false, false);
                acc[1][t] = __builtin_amdgcn_wmma_f32_16x16x32_bf16(
                    false, af[1], false, bfrag, (short)0, acc[1][t], false, false);
            }
            p ^= 1;
        }

        // ---- epilogue: bias + store (C/D layout: M = r + 8*g, N = lane&15) --
        #pragma unroll
        for (int rr = 0; rr < 2; ++rr) {
            #pragma unroll
            for (int t = 0; t < 4; ++t) {
                #pragma unroll
                for (int r = 0; r < 8; ++r) {
                    const int row = m0 + (rt0 + rr) * 16 + r + 8 * g;
                    const int c   = n0 + (ct0 + t) * 16 + hl;
                    const float o = acc[rr][t][r] + bias[c];
                    if (OUT_BF16) {
                        ((unsigned short*)Out)[(size_t)row * NOUT + c] = f2bf(o);
                    } else {
                        ((float*)Out)[(size_t)row * NOUT + c] = o;
                    }
                }
            }
        }
    }
}

// ---------------------------------------------------------------------------
// Attention core: one block per (batch-window b, head h).
// qkvb: bf16 bits [BS*49][1152] (q | k | v per row).
// ab:   bf16 bits [BS*49][384] attention output in [BS,N,C] layout.
// ---------------------------------------------------------------------------
__global__ __launch_bounds__(256)
void attn_kernel(const unsigned short* __restrict__ qkvb,
                 const float* __restrict__ mask,        // [64][49][49]
                 const float* __restrict__ bias_table,  // [169][12]
                 unsigned short* __restrict__ ab)
{
    __shared__ unsigned short lds_q [NP * HD];  // 64x32 bf16 [token][dim]
    __shared__ unsigned short lds_k [NP * HD];  // 64x32 bf16 [token][dim]
    __shared__ unsigned short lds_vt[HD * NP];  // 32x64 bf16 [dim][token]
    __shared__ float          lds_s [NP * NP];  // 64x64 f32 scores
    __shared__ unsigned short lds_p [NP * NP];  // 64x64 bf16 probs

    const int tid  = threadIdx.x;
    const int lane = tid & 31;
    const int wave = tid >> 5;
    const int b    = blockIdx.x / NHEAD;
    const int h    = blockIdx.x % NHEAD;
    const int w    = b & (NWIN - 1);

    // ---- stage q/k (row-major, async) and v (transposed); pad rows zero ----
    {
        const int row = tid >> 2;
        const int c0  = (tid & 3) * 8;
        if (row < SEQ) {
            const size_t base = (size_t)(b * SEQ + row) * (size_t)(3 * CDIM)
                              + (size_t)(h * HD + c0);
#if HAVE_ASYNC_LDS
            async_copy16(qkvb + base,        &lds_q[row * HD + c0]);   // 16B
            async_copy16(qkvb + base + CDIM, &lds_k[row * HD + c0]);   // 16B
#else
            #pragma unroll
            for (int u = 0; u < 8; ++u) {
                lds_q[row * HD + c0 + u] = qkvb[base + u];
                lds_k[row * HD + c0 + u] = qkvb[base + CDIM + u];
            }
#endif
            #pragma unroll
            for (int u = 0; u < 8; ++u)
                lds_vt[(c0 + u) * NP + row] = qkvb[base + 2 * CDIM + u];
        } else {
            #pragma unroll
            for (int u = 0; u < 8; ++u) {
                lds_q[row * HD + c0 + u] = 0;
                lds_k[row * HD + c0 + u] = 0;
                lds_vt[(c0 + u) * NP + row] = 0;
            }
        }
    }
#if HAVE_ASYNC_LDS
    async_wait0();
#endif
    __syncthreads();

    // ---- S = Q * K^T : 4x4 tiles, one 16x16x32 WMMA per tile (K == hd) ----
    {
        const int ti  = wave >> 1;
        const int tj0 = (wave & 1) * 2;
        const int g   = lane >> 4;
        v16bf_t afrag;
        {
            const int r = ti * 16 + (lane & 15);
            #pragma unroll
            for (int e = 0; e < 16; ++e) {
                const int kk = 16 * g + e;
                afrag[e] = bfbits(lds_q[r * HD + kk]);
            }
        }
        #pragma unroll
        for (int t = 0; t < 2; ++t) {
            const int tj   = tj0 + t;
            const int jrow = tj * 16 + (lane & 15);   // B[K][n] = k[n][K]: row read
            v16bf_t bfrag;
            #pragma unroll
            for (int e = 0; e < 16; ++e) {
                const int kk = e + 16 * g;
                bfrag[e] = bfbits(lds_k[jrow * HD + kk]);
            }
            v8f_t acc = {};
            acc = __builtin_amdgcn_wmma_f32_16x16x32_bf16(false, afrag, false, bfrag,
                                                          (short)0, acc, false, false);
            #pragma unroll
            for (int r = 0; r < 8; ++r) {
                const int i = ti * 16 + r + 8 * g;
                const int j = tj * 16 + (lane & 15);
                lds_s[i * NP + j] = acc[r];
            }
        }
    }
    __syncthreads();

    // ---- scale + rel-pos bias + shift mask + softmax: 4 threads per row ----
    // j < 48 handled in 12 unconditional steps; j == 48 by sub==0 tail.
    {
        const int i   = tid >> 2;     // 0..63
        const int sub = tid & 3;
        const float scale = 0.17677669529663687f;  // 32^-0.5
        float m = -1e30f;
        if (i < SEQ) {
            const int yi = i / WIN, xi = i % WIN;
            #pragma unroll 1
            for (int jj = 0; jj < 12; ++jj) {
                const int j   = sub + 4 * jj;                    // 0..47
                const int yj  = j / WIN, xj = j % WIN;
                const int idx = (yi - yj + WIN - 1) * (2 * WIN - 1)
                              + (xi - xj + WIN - 1);
                float s = lds_s[i * NP + j] * scale
                        + bias_table[idx * NHEAD + h]
                        + mask[((size_t)w * SEQ + i) * SEQ + j];
                lds_s[i * NP + j] = s;
                m = fmaxf(m, s);
            }
            if (sub == 0) {                                      // j = 48
                const int j   = 48;
                const int yj  = j / WIN, xj = j % WIN;
                const int idx = (yi - yj + WIN - 1) * (2 * WIN - 1)
                              + (xi - xj + WIN - 1);
                float s = lds_s[i * NP + j] * scale
                        + bias_table[idx * NHEAD + h]
                        + mask[((size_t)w * SEQ + i) * SEQ + j];
                lds_s[i * NP + j] = s;
                m = fmaxf(m, s);
            }
        }
        m = fmaxf(m, __shfl_xor(m, 1, 32));           // reduce over 4 lanes / row
        m = fmaxf(m, __shfl_xor(m, 2, 32));
        float sum = 0.f;
        if (i < SEQ) {
            #pragma unroll 1
            for (int jj = 0; jj < 12; ++jj) {
                const int j = sub + 4 * jj;
                const float e = __expf(lds_s[i * NP + j] - m);
                lds_s[i * NP + j] = e;
                sum += e;
            }
            if (sub == 0) {
                const float e = __expf(lds_s[i * NP + 48] - m);
                lds_s[i * NP + 48] = e;
                sum += e;
            }
        }
        sum += __shfl_xor(sum, 1, 32);
        sum += __shfl_xor(sum, 2, 32);
        const float inv = 1.f / sum;                  // rows < SEQ: sum >= 1
        #pragma unroll
        for (int jj = 0; jj < 16; ++jj) {
            const int j = sub + 4 * jj;               // covers 0..63
            const float pv = (i < SEQ && j < SEQ) ? lds_s[i * NP + j] * inv : 0.f;
            lds_p[i * NP + j] = f2bf(pv);
        }
    }
    __syncthreads();

    // ---- O = P * V : 64x32 out, 8 tiles (4x2), 2 chained WMMAs each (K=64) --
    {
        const int mi = wave >> 1;
        const int nj = wave & 1;
        const int g  = lane >> 4;
        v8f_t acc = {};
        #pragma unroll
        for (int ks = 0; ks < 2; ++ks) {
            v16bf_t afrag, bfrag;
            const int r = mi * 16 + (lane & 15);
            #pragma unroll
            for (int e = 0; e < 16; ++e) {
                const int kk = ks * 32 + 16 * g + e;
                afrag[e] = bfbits(lds_p[r * NP + kk]);
            }
            const int n_ = nj * 16 + (lane & 15);     // dim 0..31
            #pragma unroll
            for (int e = 0; e < 16; ++e)              // contiguous in lds_vt
                bfrag[e] = bfbits(lds_vt[n_ * NP + ks * 32 + 16 * g + e]);
            acc = __builtin_amdgcn_wmma_f32_16x16x32_bf16(false, afrag, false, bfrag,
                                                          (short)0, acc, false, false);
        }
        #pragma unroll
        for (int r = 0; r < 8; ++r) {
            const int i = mi * 16 + r + 8 * g;
            const int c = nj * 16 + (lane & 15);
            if (i < SEQ)
                ab[(size_t)(b * SEQ + i) * CDIM + (size_t)(h * HD + c)] = f2bf(acc[r]);
        }
    }
}

// ---------------------------------------------------------------------------
// Launch: QKV GEMM -> attention -> proj GEMM (all on `stream`).
// Workspace: qkvb bf16 [200704][1152] (462 MB) | ab bf16 [200704][384] (154 MB)
// ---------------------------------------------------------------------------
extern "C" void kernel_launch(void* const* d_in, const int* in_sizes, int n_in,
                              void* d_out, int out_size, void* d_ws, size_t ws_size,
                              hipStream_t stream) {
    const float* x          = (const float*)d_in[0];
    const float* mask       = (const float*)d_in[1];
    const float* qkv_w      = (const float*)d_in[2];
    const float* qkv_b      = (const float*)d_in[3];
    const float* proj_w     = (const float*)d_in[4];
    const float* proj_b     = (const float*)d_in[5];
    const float* bias_table = (const float*)d_in[6];
    (void)in_sizes; (void)n_in; (void)out_size; (void)ws_size;

    unsigned short* qkvb = (unsigned short*)d_ws;
    unsigned short* ab   = (unsigned short*)((char*)d_ws +
                           (size_t)MROWS * (size_t)(3 * CDIM) * sizeof(unsigned short));

    // 1) qkv = x @ qkv_w + qkv_b  (fp32 in, bf16 out), NT = 1152/128 = 9
    gemm_bias_kernel<false, true, 9><<<MROWS / 128, 256, GEMM_LDS_BYTES, stream>>>(
        x, qkv_w, qkv_b, qkvb);

    // 2) windowed attention per (batch, head)
    attn_kernel<<<BATCH * NHEAD, 256, 0, stream>>>(qkvb, mask, bias_table, ab);

    // 3) out = attn_out @ proj_w + proj_b  (bf16 in, fp32 out), NT = 384/128 = 3
    gemm_bias_kernel<true, false, 3><<<MROWS / 128, 256, GEMM_LDS_BYTES, stream>>>(
        ab, proj_w, proj_b, (float*)d_out);
}
